// Layer_38474317037991
// MI455X (gfx1250) — matrix operand order repaired and checked by reading, not compiled, not run on hardware
//
#include <hip/hip_runtime.h>
#include <hip/hip_bf16.h>
#include <math.h>

typedef __attribute__((ext_vector_type(16))) __bf16 v16bf;
typedef __attribute__((ext_vector_type(2)))  __bf16 v2bf;
typedef __attribute__((ext_vector_type(8)))  float  v8f;

#define N_NODE  50000
#define NEDGE   500000
#define D_IN    256
#define NHEAD   4
#define D_HID   64
#define HDTOT   256      // NHEAD * D_HID
#define D_CAT   320      // D_HID + D_IN
#define D_OUT   128
#define NEG_SLOPE 0.2f

// ---------------------------------------------------------------------------
// helpers
// ---------------------------------------------------------------------------
__device__ __forceinline__ void atomic_max_f32(float* addr, float val) {
  // monotonic float max via integer atomics (init must be -inf)
  if (val >= 0.0f) atomicMax((int*)addr, __float_as_int(val));
  else             atomicMin((unsigned int*)addr, __float_as_uint(val));
}

__global__ void fill_f32_kernel(float* __restrict__ p, float v, long long n) {
  long long i = (long long)blockIdx.x * blockDim.x + threadIdx.x;
  long long stride = (long long)gridDim.x * blockDim.x;
  for (; i < n; i += stride) p[i] = v;
}

// A-fragment K index for element pair starting at 2j (16-bit A 16x32 layout):
// element e -> K=2*kb+e (contiguous, matches B)   (kb = (lane>>4)*8)
__device__ __forceinline__ int a_pair_k(int kb, int j) {
  return 2 * kb + 2 * j;
}

// ---------------------------------------------------------------------------
// Kernel 1: P = X @ fc_w  (fp32 -> bf16 WMMA, fp32 accum)
// Fused: elr[n,h] = sum_d P[n, h*64+d] * attn[h,d]
// Block: 256 threads (8 waves). Tile: 64 rows x 256 cols.
// Wave w owns col slices {2w, 2w+1}; 4 row subtiles => 8 WMMA accums / wave.
// LDS is staged FRAGMENT-MAJOR so each lane reads its v16bf with one 32B load.
// GUARD=true only for the single overhang block (rows >= N clamped/skipped).
// ---------------------------------------------------------------------------
template <bool GUARD>
__global__ __launch_bounds__(256, 1)
void fc_gemm_kernel(const float* __restrict__ X,     // [N,256]
                    const float* __restrict__ W,     // [256,256]
                    const float* __restrict__ attn,  // [4,64]
                    __bf16*      __restrict__ P,     // [N,256] or null
                    float*       __restrict__ elr,   // [N,4] pre-zeroed
                    int N, int row_base) {
  __shared__ alignas(32) __bf16 sAf[4 * 512];   // [row-subtile][lane*16+e]
  __shared__ alignas(32) __bf16 sBf[16 * 512];  // [col-slice][lane*16+e]
  const int t    = threadIdx.x;
  const int wave = t >> 5;
  const int lane = t & 31;
  const int row0 = row_base + blockIdx.x * 64;

  v8f acc[8] = {};   // [rs*2 + cs]

  for (int k0 = 0; k0 < D_IN; k0 += 32) {
    // ---- stage A tile 64x32: 1024 bf16-pairs, 4 per thread, float2 loads ----
#pragma unroll
    for (int i = 0; i < 4; ++i) {
      int p  = t + i * 256;
      int j  = p & 7;             // pair index within lane fragment
      int lp = (p >> 3) & 31;     // fragment lane position
      int st = p >> 8;            // row subtile
      int kb = (lp >> 4) * 8;
      int K0 = a_pair_k(kb, j);
      int r  = row0 + st * 16 + (lp & 15);
      if (GUARD) r = r < N ? r : N - 1;   // clamp (stores guarded later)
      float2 xv = *(const float2*)&X[(long long)r * D_IN + k0 + K0];
      v2bf pk = { (__bf16)xv.x, (__bf16)xv.y };
      *(v2bf*)&sAf[st * 512 + lp * 16 + 2 * j] = pk;
    }
    // ---- stage B tile 32x256: 4096 pairs, 16 per thread (coalesced cols) ----
#pragma unroll
    for (int i = 0; i < 16; ++i) {
      int q    = t + i * 256;
      int col  = q & 255;
      int r8   = q >> 8;          // 0..15
      int half = r8 & 1;
      int j    = r8 >> 1;
      int lp   = (half << 4) | (col & 15);
      int slic = col >> 4;
      int K    = half * 16 + 2 * j;
      float w0 = W[(long long)(k0 + K) * HDTOT + col];
      float w1 = W[(long long)(k0 + K + 1) * HDTOT + col];
      v2bf pk = { (__bf16)w0, (__bf16)w1 };
      *(v2bf*)&sBf[slic * 512 + lp * 16 + 2 * j] = pk;
    }
    __syncthreads();

    v16bf af[4], bf[2];
#pragma unroll
    for (int rs = 0; rs < 4; ++rs)
      af[rs] = *(const v16bf*)&sAf[rs * 512 + lane * 16];
#pragma unroll
    for (int cs = 0; cs < 2; ++cs)
      bf[cs] = *(const v16bf*)&sBf[(2 * wave + cs) * 512 + lane * 16];

#pragma unroll
    for (int rs = 0; rs < 4; ++rs)
#pragma unroll
      for (int cs = 0; cs < 2; ++cs)
        acc[rs * 2 + cs] = __builtin_amdgcn_wmma_f32_16x16x32_bf16(
            false, af[rs], false, bf[cs], (short)0, acc[rs * 2 + cs], false, false);
    __syncthreads();
  }

  // ---- epilogue: P store (bf16) + fused attention dot product ----
  const int hi = lane >> 4;
  const int nc = lane & 31 & 15;
#pragma unroll
  for (int rs = 0; rs < 4; ++rs) {
#pragma unroll
    for (int cs = 0; cs < 2; ++cs) {
      const int col = (2 * wave + cs) * 16 + nc;   // 0..255
      const int h   = col >> 6;
      const int d   = col & 63;
      const float al = attn[h * D_HID + d];
      v8f a = acc[rs * 2 + cs];
      if (P) {
#pragma unroll
        for (int i = 0; i < 8; ++i) {
          int r = row0 + rs * 16 + i + hi * 8;
          if (!GUARD || r < N) P[(long long)r * HDTOT + col] = (__bf16)a[i];
        }
      }
#pragma unroll
      for (int i = 0; i < 8; ++i) {
        float p = a[i] * al;
        p += __shfl_xor(p, 1);
        p += __shfl_xor(p, 2);
        p += __shfl_xor(p, 4);
        p += __shfl_xor(p, 8);
        if (nc == 0) {
          int r = row0 + rs * 16 + i + hi * 8;
          if (!GUARD || r < N) atomicAdd(&elr[(long long)r * NHEAD + h], p);
        }
      }
    }
  }
}

// ---------------------------------------------------------------------------
// Kernel 2: e = leaky_relu(el[src]+er[dst]); segment max into m[dst,h]
// ---------------------------------------------------------------------------
__global__ void edge_logit_max_kernel(const int* __restrict__ src,
                                      const int* __restrict__ dst,
                                      const float* __restrict__ el,  // [Ns,4]
                                      const float* __restrict__ er,  // [Nd,4]
                                      float* __restrict__ ebuf,      // [E,4]
                                      float* __restrict__ mbuf,      // [Nd,4] = -inf
                                      int E) {
  int i = blockIdx.x * blockDim.x + threadIdx.x;
  if (i >= E) return;
  int s = src[i], d = dst[i];
  float4 l = ((const float4*)el)[s];
  float4 r = ((const float4*)er)[d];
  float4 e;
  e.x = l.x + r.x; e.x = e.x > 0.f ? e.x : NEG_SLOPE * e.x;
  e.y = l.y + r.y; e.y = e.y > 0.f ? e.y : NEG_SLOPE * e.y;
  e.z = l.z + r.z; e.z = e.z > 0.f ? e.z : NEG_SLOPE * e.z;
  e.w = l.w + r.w; e.w = e.w > 0.f ? e.w : NEG_SLOPE * e.w;
  ((float4*)ebuf)[i] = e;
  atomic_max_f32(&mbuf[d * 4 + 0], e.x);
  atomic_max_f32(&mbuf[d * 4 + 1], e.y);
  atomic_max_f32(&mbuf[d * 4 + 2], e.z);
  atomic_max_f32(&mbuf[d * 4 + 3], e.w);
}

// ---------------------------------------------------------------------------
// Kernel 3: ex = exp(e - m[dst]) (in place); den[dst,h] += ex
// ---------------------------------------------------------------------------
__global__ void edge_exp_den_kernel(const int* __restrict__ dst,
                                    float* __restrict__ ebuf,       // [E,4] in/out
                                    const float* __restrict__ mbuf, // [Nd,4]
                                    float* __restrict__ den,        // [Nd,4] zeroed
                                    int E) {
  int i = blockIdx.x * blockDim.x + threadIdx.x;
  if (i >= E) return;
  int d = dst[i];
  float4 m = ((const float4*)mbuf)[d];
  float4 e = ((const float4*)ebuf)[i];
  e.x = __expf(e.x - m.x);
  e.y = __expf(e.y - m.y);
  e.z = __expf(e.z - m.z);
  e.w = __expf(e.w - m.w);
  ((float4*)ebuf)[i] = e;
  atomicAdd(&den[d * 4 + 0], e.x);
  atomicAdd(&den[d * 4 + 1], e.y);
  atomicAdd(&den[d * 4 + 2], e.z);
  atomicAdd(&den[d * 4 + 3], e.w);
}

// ---------------------------------------------------------------------------
// Kernel 4: agg[dst, :] += alpha[e,h] * P[src, :]   (one 256-wide block / edge)
// ---------------------------------------------------------------------------
__global__ __launch_bounds__(256)
void aggregate_kernel(const int* __restrict__ src,
                      const int* __restrict__ dst,
                      const float* __restrict__ ex,   // [E,4]
                      const float* __restrict__ den,  // [Nd,4]
                      const __bf16* __restrict__ P,   // [Ns,256]
                      float* __restrict__ agg) {      // [Nd,256] zeroed
  int e = blockIdx.x;
  int t = threadIdx.x;          // 0..255 == h*64 + d
  int s = src[e], d = dst[e];
  int h = t >> 6;
  float dn = den[d * 4 + h];
  float alpha = ex[e * 4 + h] / (dn == 0.0f ? 1.0f : dn);
  float v = (float)P[(long long)s * HDTOT + t];
  atomicAdd(&agg[(long long)d * HDTOT + t], alpha * v);
}

// ---------------------------------------------------------------------------
// Kernel 5: z = relu(concat(mean_h(agg+bias), h_dst) @ W_w + W_b); L2 normalize
// Block: 256 threads (8 waves), tile 16 rows x 128 cols; K=320 in steps of 32.
// Fragment-major LDS staging; N is an exact multiple of 16 -> no guards.
// ---------------------------------------------------------------------------
__global__ __launch_bounds__(256, 1)
void head_kernel(const float* __restrict__ agg,      // [N,4,64]
                 const float* __restrict__ gat_bias, // [256]
                 const float* __restrict__ hdst,     // [N,256]
                 const float* __restrict__ Ww,       // [320,128]
                 const float* __restrict__ Wb,       // [128]
                 float* __restrict__ out,            // [N,128]
                 int N) {
  __shared__ alignas(32) __bf16 sAf[512];       // [lane*16+e]
  __shared__ alignas(32) __bf16 sBf[8 * 512];   // [col-slice][lane*16+e]
  __shared__ float  sZ[16][128];
  __shared__ float  sRN[16];
  const int t    = threadIdx.x;
  const int wave = t >> 5;
  const int lane = t & 31;
  const int row0 = blockIdx.x * 16;

  v8f acc = {};
  for (int k0 = 0; k0 < D_CAT; k0 += 32) {
    // ---- A tile 16x32, fragment-major: 256 pairs, 1 per thread ----
    {
      int j  = t & 7;
      int lp = (t >> 3) & 31;
      int kb = (lp >> 4) * 8;
      int K0 = a_pair_k(kb, j);
      int n  = row0 + (lp & 15);
      int gk = k0 + K0;           // gk and gk+1 are always in the same region
      float v0, v1;
      if (gk < D_HID) {
        float s0 = 0.f, s1 = 0.f;
#pragma unroll
        for (int h = 0; h < NHEAD; ++h) {
          s0 += agg[(long long)n * HDTOT + h * D_HID + gk]     + gat_bias[h * D_HID + gk];
          s1 += agg[(long long)n * HDTOT + h * D_HID + gk + 1] + gat_bias[h * D_HID + gk + 1];
        }
        v0 = 0.25f * s0; v1 = 0.25f * s1;
      } else {
        float2 hv = *(const float2*)&hdst[(long long)n * D_IN + (gk - D_HID)];
        v0 = hv.x; v1 = hv.y;
      }
      v2bf pk = { (__bf16)v0, (__bf16)v1 };
      *(v2bf*)&sAf[lp * 16 + 2 * j] = pk;
    }
    // ---- B tile 32x128: 2048 pairs, 8 per thread (coalesced cols) ----
#pragma unroll
    for (int i = 0; i < 8; ++i) {
      int q    = t + i * 256;
      int col  = q & 127;
      int r8   = q >> 7;          // 0..15
      int half = r8 & 1;
      int j    = r8 >> 1;
      int lp   = (half << 4) | (col & 15);
      int slic = col >> 4;
      int K    = half * 16 + 2 * j;
      float w0 = Ww[(long long)(k0 + K) * D_OUT + col];
      float w1 = Ww[(long long)(k0 + K + 1) * D_OUT + col];
      v2bf pk = { (__bf16)w0, (__bf16)w1 };
      *(v2bf*)&sBf[slic * 512 + lp * 16 + 2 * j] = pk;
    }
    __syncthreads();

    v16bf a = *(const v16bf*)&sAf[lane * 16];
    v16bf b = *(const v16bf*)&sBf[wave * 512 + lane * 16];
    acc = __builtin_amdgcn_wmma_f32_16x16x32_bf16(false, a, false, b,
                                                  (short)0, acc, false, false);
    __syncthreads();
  }

  // bias + relu -> stage full 16x128 tile in LDS
  const int hi  = lane >> 4;
  const int nc  = lane & 15;
  const int col = wave * 16 + nc;
  const float wb = Wb[col];
#pragma unroll
  for (int i = 0; i < 8; ++i) {
    float z = acc[i] + wb;
    sZ[i + hi * 8][col] = fmaxf(z, 0.0f);
  }
  __syncthreads();

  // row L2 norms: 8 threads per row, 16 cols each, shfl-reduce
  if (t < 128) {
    int r  = t >> 3;
    int c0 = (t & 7) * 16;
    float s = 0.f;
#pragma unroll
    for (int j = 0; j < 16; ++j) { float z = sZ[r][c0 + j]; s += z * z; }
    s += __shfl_xor(s, 1);
    s += __shfl_xor(s, 2);
    s += __shfl_xor(s, 4);
    if ((t & 7) == 0) {
      float nn = sqrtf(s);
      sRN[r] = (nn == 0.0f) ? 1.0f : 1.0f / nn;
    }
  }
  __syncthreads();

  for (int i = t; i < 16 * 128; i += 256) {
    int r = i >> 7, c = i & 127;
    out[(long long)(row0 + r) * D_OUT + c] = sZ[r][c] * sRN[r];
  }
}

// ---------------------------------------------------------------------------
// launch
// ---------------------------------------------------------------------------
static inline void launch_fc(const float* X, const float* W, const float* attn,
                             __bf16* P, float* elr, hipStream_t stream) {
  const int full_blocks = N_NODE / 64;          // 781 unguarded
  const int rem_rows    = N_NODE - full_blocks * 64;  // 16
  fc_gemm_kernel<false><<<full_blocks, 256, 0, stream>>>(X, W, attn, P, elr,
                                                         N_NODE, 0);
  if (rem_rows > 0)
    fc_gemm_kernel<true><<<1, 256, 0, stream>>>(X, W, attn, P, elr,
                                                N_NODE, full_blocks * 64);
}

extern "C" void kernel_launch(void* const* d_in, const int* in_sizes, int n_in,
                              void* d_out, int out_size, void* d_ws, size_t ws_size,
                              hipStream_t stream) {
  const float* h_src_user = (const float*)d_in[0];
  const float* h_src_item = (const float*)d_in[1];
  const float* h_dst_user = (const float*)d_in[2];
  const float* h_dst_item = (const float*)d_in[3];
  const int*   u2i_src    = (const int*)d_in[4];
  const int*   u2i_dst    = (const int*)d_in[5];
  const int*   i2u_src    = (const int*)d_in[6];
  const int*   i2u_dst    = (const int*)d_in[7];
  const float* fc_w       = (const float*)d_in[8];
  const float* attn_l     = (const float*)d_in[9];
  const float* attn_r     = (const float*)d_in[10];
  const float* gat_bias   = (const float*)d_in[11];
  const float* W_w        = (const float*)d_in[12];
  const float* W_b        = (const float*)d_in[13];

  const long long P_bytes   = (long long)N_NODE * HDTOT * sizeof(__bf16); // 25.6MB
  const long long E4_bytes  = (long long)NEDGE * 4 * sizeof(float);       // 8MB
  const long long NH_bytes  = (long long)N_NODE * NHEAD * sizeof(float);  // 0.8MB
  const long long AGG_bytes = (long long)N_NODE * HDTOT * sizeof(float);  // 51.2MB

  char* w = (char*)d_ws;
  __bf16* P_u   = (__bf16*)w; w += P_bytes;
  __bf16* P_i   = (__bf16*)w; w += P_bytes;
  float*  e_u2i = (float*)w;  w += E4_bytes;
  float*  e_i2u = (float*)w;  w += E4_bytes;
  // ---- contiguous zero-init region ----
  float*  zbase = (float*)w;
  float*  el_u  = (float*)w;  w += NH_bytes;
  float*  el_i  = (float*)w;  w += NH_bytes;
  float*  er_u  = (float*)w;  w += NH_bytes;
  float*  er_i  = (float*)w;  w += NH_bytes;
  float*  den_i = (float*)w;  w += NH_bytes;
  float*  den_u = (float*)w;  w += NH_bytes;
  float*  agg_i = (float*)w;  w += AGG_bytes;
  float*  agg_u = (float*)w;  w += AGG_bytes;
  const long long zero_n = (6LL * NH_bytes + 2LL * AGG_bytes) / (long long)sizeof(float);
  // ---- -inf init region (segment max) ----
  float*  m_i   = (float*)w;  w += NH_bytes;
  float*  m_u   = (float*)w;  w += NH_bytes;
  const long long minf_n = 2LL * NH_bytes / (long long)sizeof(float);

  fill_f32_kernel<<<4096, 256, 0, stream>>>(zbase, 0.0f, zero_n);
  fill_f32_kernel<<<512, 256, 0, stream>>>(m_i, -INFINITY, minf_n);

  // src projections (store P, el with attn_l); dst projections (er with attn_r)
  launch_fc(h_src_user, fc_w, attn_l, P_u, el_u, stream);
  launch_fc(h_src_item, fc_w, attn_l, P_i, el_i, stream);
  launch_fc(h_dst_item, fc_w, attn_r, (__bf16*)nullptr, er_i, stream);
  launch_fc(h_dst_user, fc_w, attn_r, (__bf16*)nullptr, er_u, stream);

  const int eblocks = (NEDGE + 255) / 256;
  // graph u2i: src=user, dst=item
  edge_logit_max_kernel<<<eblocks, 256, 0, stream>>>(u2i_src, u2i_dst, el_u, er_i, e_u2i, m_i, NEDGE);
  // graph i2u: src=item, dst=user
  edge_logit_max_kernel<<<eblocks, 256, 0, stream>>>(i2u_src, i2u_dst, el_i, er_u, e_i2u, m_u, NEDGE);

  edge_exp_den_kernel<<<eblocks, 256, 0, stream>>>(u2i_dst, e_u2i, m_i, den_i, NEDGE);
  edge_exp_den_kernel<<<eblocks, 256, 0, stream>>>(i2u_dst, e_i2u, m_u, den_u, NEDGE);

  aggregate_kernel<<<NEDGE, 256, 0, stream>>>(u2i_src, u2i_dst, e_u2i, den_i, P_u, agg_i);
  aggregate_kernel<<<NEDGE, 256, 0, stream>>>(i2u_src, i2u_dst, e_i2u, den_u, P_i, agg_u);

  float* out_user = (float*)d_out;
  float* out_item = out_user + (long long)N_NODE * D_OUT;
  const int head_blocks = N_NODE / 16;   // 3125, exact
  head_kernel<<<head_blocks, 256, 0, stream>>>(agg_u, gat_bias, h_dst_user, W_w, W_b, out_user, N_NODE);
  head_kernel<<<head_blocks, 256, 0, stream>>>(agg_i, gat_bias, h_dst_item, W_w, W_b, out_item, N_NODE);

  (void)in_sizes; (void)n_in; (void)out_size; (void)ws_size;
}
